// MultiHeadAttention_ProbSparse_33758442946701
// MI455X (gfx1250) — matrix operand order repaired and checked by reading, not compiled, not run on hardware
//
#include <hip/hip_runtime.h>
#include <hip/hip_bf16.h>

// ---------------------------------------------------------------------------
// ProbSparse MHA with L_Q == 1  ==>  exact algebraic collapse:
//   out = concat_h( softmax( (k@Wk+bk)·(qh)ᵀ /8 ) · (v@Wv+bv) ) @ Wo + bo
// Folding Wk into the query (u = Wkᵀ-slice · qh) and deferring Wv until after
// the attn·V reduction turns 275 GFLOP of projections into ~4.5 GFLOP, making
// this HBM-bound on one streaming pass of k,v (536 MB ≈ 23 us at 23.3 TB/s).
// Both Lk-sized reductions use v_wmma_f32_16x16x32_bf16 (f32 accumulate).
// The attn·V kernel streams V through the Tensor Data Mover (TDM) into LDS,
// double-buffered, overlapping HBM fetch with convert+WMMA.
// index_sample is provably dead (top_k over a length-1 axis) and ignored.
// ---------------------------------------------------------------------------

#define HIDDEN 1024
#define HEADS  16
#define D_HEAD 64
#define BSZ    16
#define LK     4096

typedef __bf16 bf16_t;
typedef __attribute__((ext_vector_type(16))) __bf16        v16bf;
typedef __attribute__((ext_vector_type(8)))  __bf16        v8bf;
typedef __attribute__((ext_vector_type(4)))  __bf16        v4bf;
typedef __attribute__((ext_vector_type(8)))  float         v8f;
typedef __attribute__((ext_vector_type(4)))  float         v4f;
typedef __attribute__((ext_vector_type(4)))  unsigned int  v4u;
typedef __attribute__((ext_vector_type(8)))  int           v8i;
typedef __attribute__((ext_vector_type(4)))  int           v4i;

#if defined(__has_builtin)
#if __has_builtin(__builtin_amdgcn_tensor_load_to_lds) && \
    __has_builtin(__builtin_amdgcn_s_wait_tensorcnt)
#define HAVE_TDM 1
#endif
#endif
#ifndef HAVE_TDM
#define HAVE_TDM 0
#endif

// ---------------- P0a: qh = q@Wq + bq ; zero the t accumulator --------------
__global__ void __launch_bounds__(256)
prep_qh_zero_t(const float* __restrict__ q, const float* __restrict__ Wq,
               const float* __restrict__ bq, float* __restrict__ qh,
               float* __restrict__ t) {
  const int gid = blockIdx.x * 256 + threadIdx.x;       // 262144 threads
  t[gid] = 0.0f;                                        // t is B*H*HIDDEN
  if (gid < BSZ * HIDDEN) {
    const int b = gid >> 10, j = gid & 1023;
    const float* qp = q + b * HIDDEN;
    float acc = bq[j];
    for (int c = 0; c < HIDDEN; ++c) acc += qp[c] * Wq[c * HIDDEN + j];
    qh[gid] = acc;
  }
}

// ---------------- P0b: u[b,h,c] = sum_d Wk[c,h*64+d]*qh[b,h*64+d] (bf16) ----
//                   sbias[b,h]  = sum_d bk[h*64+d]*qh[b,h*64+d]
__global__ void __launch_bounds__(256)
prep_u(const float* __restrict__ Wk, const float* __restrict__ bk,
       const float* __restrict__ qh, bf16_t* __restrict__ u,
       float* __restrict__ sbias) {
  const int gid = blockIdx.x * 256 + threadIdx.x;       // 262144 = B*H*HIDDEN
  const int c = gid & 1023;
  const int h = (gid >> 10) & 15;
  const int b = gid >> 14;
  const float* qp = qh + b * HIDDEN + h * D_HEAD;
  const float* wp = Wk + (size_t)c * HIDDEN + h * D_HEAD;
  float acc = 0.0f;
  for (int d = 0; d < D_HEAD; ++d) acc += wp[d] * qp[d];
  u[gid] = (__bf16)acc;
  if (c == 0) {
    const float* bkp = bk + h * D_HEAD;
    float sb = 0.0f;
    for (int d = 0; d < D_HEAD; ++d) sb += bkp[d] * qp[d];
    sbias[b * HEADS + h] = sb;
  }
}

// ---------------- S: scores[b,h,l] = (k[b,l,:]·u[b,h,:] + sbias)*0.125 ------
// Per workgroup: 128 keys x 16 heads; 8 waves, one 16x16 D-tile per wave.
// A (16x32 bf16, M=key): lane m=lane&15, g=lane>>4; elems e -> K=kc+16g+e,
// two contiguous 8-float runs per lane (lane pair
// l,l+16 together consume a full 128B line of the same k row).
// B (32x16 bf16, N=head): lane n=lane&15, elems e -> K=kc+16g+e (contiguous).
__global__ void __launch_bounds__(256)
scores_kernel(const float* __restrict__ k, const bf16_t* __restrict__ u,
              const float* __restrict__ sbias, float* __restrict__ s) {
  const int b     = blockIdx.x;                 // 16
  const int ltile = blockIdx.y;                 // 32 tiles of 128 keys
  const int wave  = threadIdx.x >> 5;
  const int lane  = threadIdx.x & 31;
  const int g     = lane >> 4;
  const int m     = lane & 15;                  // A row (key) and B col (head)
  const int lbase = ltile * 128 + wave * 16;

  const float*  krow = k + ((size_t)b * LK + (lbase + m)) * HIDDEN;
  const bf16_t* urow = u + ((size_t)(b * HEADS + m)) * HIDDEN;

  v8f acc = {};
  for (int kc = 0; kc < HIDDEN; kc += 32) {
    __builtin_prefetch(krow + kc + 256, 0, 1);          // global_prefetch_b8
    const float* pa0 = krow + kc + 16 * g;
    const float* pa1 = krow + kc + 16 * g + 8;
    v4f a0 = *(const v4f*)(pa0);
    v4f a1 = *(const v4f*)(pa0 + 4);
    v4f a2 = *(const v4f*)(pa1);
    v4f a3 = *(const v4f*)(pa1 + 4);
    v16bf A;
#pragma unroll
    for (int i = 0; i < 4; ++i) {
      A[i]      = (__bf16)a0[i];
      A[4 + i]  = (__bf16)a1[i];
      A[8 + i]  = (__bf16)a2[i];
      A[12 + i] = (__bf16)a3[i];
    }
    v16bf Bf = *(const v16bf*)(urow + kc + 16 * g);
    acc = __builtin_amdgcn_wmma_f32_16x16x32_bf16(false, A, false, Bf,
                                                  (short)0, acc, false, false);
  }
  // D layout: VGPR r, lanes0-15 -> M=r, lanes16-31 -> M=8+r; N = m (head).
  const float sb = sbias[b * HEADS + m];
  float* srow = s + ((size_t)(b * HEADS + m)) * LK + lbase + 8 * g;
#pragma unroll
  for (int r = 0; r < 8; ++r) srow[r] = (acc[r] + sb) * 0.125f;
}

// ---------------- SM: row softmax over Lk, emit attn as bf16 ----------------
__global__ void __launch_bounds__(256)
softmax_kernel(const float* __restrict__ s, bf16_t* __restrict__ attn) {
  __shared__ float red[256];
  const int bh = blockIdx.x;                    // 256 = B*H
  const int t  = threadIdx.x;
  const float* row = s + (size_t)bh * LK;
  float vals[16];
  float mx = -1e30f;
#pragma unroll
  for (int i = 0; i < 16; ++i) { vals[i] = row[t + 256 * i]; mx = fmaxf(mx, vals[i]); }
  red[t] = mx; __syncthreads();
  for (int off = 128; off > 0; off >>= 1) {
    if (t < off) red[t] = fmaxf(red[t], red[t + off]);
    __syncthreads();
  }
  mx = red[0]; __syncthreads();
  float sum = 0.0f;
#pragma unroll
  for (int i = 0; i < 16; ++i) { vals[i] = __expf(vals[i] - mx); sum += vals[i]; }
  red[t] = sum; __syncthreads();
  for (int off = 128; off > 0; off >>= 1) {
    if (t < off) red[t] += red[t + off];
    __syncthreads();
  }
  const float inv = 1.0f / red[0];
  bf16_t* arow = attn + (size_t)bh * LK;
#pragma unroll
  for (int i = 0; i < 16; ++i) arow[t + 256 * i] = (__bf16)(vals[i] * inv);
}

#if HAVE_TDM
// Issue a TDM load of a 128(cols,f32) x 32(rows) tile of v[b] into LDS.
// D# per CDNA5 ISA ch.8: group0 = {count=1 | lds_addr | global_addr | type=2},
// group1 = {data_size=4B, tensor_dim0=1024, tensor_dim1=4096, tile=128x32,
//           tensor_dim0_stride=1024}. Groups 2/3 zero (<=2D tensor).
__device__ __forceinline__ void tdm_load_v_tile(const float* gptr,
                                                unsigned int lds_addr) {
  const unsigned long long ga = (unsigned long long)(uintptr_t)gptr;
  v4u g0;
  g0[0] = 1u;                                           // count=1, user D#
  g0[1] = lds_addr;                                     // LDS byte address
  g0[2] = (unsigned int)ga;                             // global_addr[31:0]
  g0[3] = (unsigned int)((ga >> 32) & 0x01FFFFFFu) | 0x80000000u; // addr|type=2
  v8i g1;
  g1[0] = 2 << 16;                                      // data_size = 4 bytes
  g1[1] = (int)(1024u << 16);                           // tensor_dim0 = 1024
  g1[2] = (int)(4096u << 16);                           // tensor_dim1 = 4096
  g1[3] = (int)(128u  << 16);                           // tile_dim0  = 128
  g1[4] = 32;                                           // tile_dim1  = 32
  g1[5] = 1024;                                         // tensor_dim0_stride
  g1[6] = 0;
  g1[7] = 0;
  v4i gz4 = {0, 0, 0, 0};
#if __clang_major__ >= 23
  v8i gz8 = {0, 0, 0, 0, 0, 0, 0, 0};
  __builtin_amdgcn_tensor_load_to_lds(g0, g1, gz4, gz4, gz8, 0);
#else
  __builtin_amdgcn_tensor_load_to_lds(g0, g1, gz4, gz4, 0);
#endif
}
#endif

// ---------------- T: t[b,h,c] += attn[b,h,:]·v[b,:,c]  (WMMA, K split 4x) ---
// M=16 heads, N=128 cols per WG (16/wave), K chunk of 1024 per blockIdx.z.
// V tiles (32 x 128 f32) stream via double-buffered TDM into LDS; threads
// convert to bf16 and transpose into ldsT[col][row] with packed ds_store_b64
// (each thread owns a 4-row x 4-col block), then each lane reads its WMMA
// B-fragment as one contiguous 32B run.
__global__ void __launch_bounds__(256)
av_kernel(const bf16_t* __restrict__ attn, const float* __restrict__ v,
          float* __restrict__ t) {
  __shared__ __align__(32) bf16_t ldsT[128 * 32];       // [colLocal][row] 8KB
#if HAVE_TDM
  __shared__ __align__(16) float bufF[2][32 * 128];     // f32 V tiles, 2x16KB
#endif
  const int b     = blockIdx.x;                         // 16
  const int ntile = blockIdx.y;                         // 8 -> 128 cols
  const int ks    = blockIdx.z;                         // 4 -> 1024 keys
  const int wave  = threadIdx.x >> 5;
  const int lane  = threadIdx.x & 31;
  const int g     = lane >> 4;
  const int m     = lane & 15;                          // A row (head), B col
  const int cbase = ntile * 128;
  const int lbase = ks * 1024;

  const bf16_t* arow = attn + ((size_t)(b * HEADS + m)) * LK + lbase;
  const float*  vblk = v + ((size_t)b * LK + lbase) * HIDDEN + cbase;

  // staging decomposition: thread owns rows 4p..4p+3, cols 4cq..4cq+3
  const int p  = threadIdx.x >> 5;                      // 0..7
  const int cq = threadIdx.x & 31;                      // 0..31

#if HAVE_TDM
  if (wave == 0) tdm_load_v_tile(vblk, (unsigned int)(uintptr_t)&bufF[0][0]);
#endif

  v8f acc = {};
  for (int it = 0; it < 32; ++it) {
    const int kc  = it * 32;
    const int cur = it & 1;
#if HAVE_TDM
    if (wave == 0) {
      if (it + 1 < 32) {
        tdm_load_v_tile(vblk + (size_t)(kc + 32) * HIDDEN,
                        (unsigned int)(uintptr_t)&bufF[cur ^ 1][0]);
        __builtin_amdgcn_s_wait_tensorcnt(1);           // tile `it` resident
      } else {
        __builtin_amdgcn_s_wait_tensorcnt(0);
      }
    }
    __syncthreads();            // bufF[cur] ready; prev ldsT reads done
    v4f r0 = *(const v4f*)&bufF[cur][(4 * p + 0) * 128 + 4 * cq];
    v4f r1 = *(const v4f*)&bufF[cur][(4 * p + 1) * 128 + 4 * cq];
    v4f r2 = *(const v4f*)&bufF[cur][(4 * p + 2) * 128 + 4 * cq];
    v4f r3 = *(const v4f*)&bufF[cur][(4 * p + 3) * 128 + 4 * cq];
#else
    (void)cur;
    v4f r0 = *(const v4f*)(vblk + (size_t)(kc + 4 * p + 0) * HIDDEN + 4 * cq);
    v4f r1 = *(const v4f*)(vblk + (size_t)(kc + 4 * p + 1) * HIDDEN + 4 * cq);
    v4f r2 = *(const v4f*)(vblk + (size_t)(kc + 4 * p + 2) * HIDDEN + 4 * cq);
    v4f r3 = *(const v4f*)(vblk + (size_t)(kc + 4 * p + 3) * HIDDEN + 4 * cq);
    __syncthreads();            // prev ldsT reads done
#endif
#pragma unroll
    for (int j = 0; j < 4; ++j) {                       // 4x ds_store_b64
      v4bf pk;
      pk[0] = (__bf16)r0[j];
      pk[1] = (__bf16)r1[j];
      pk[2] = (__bf16)r2[j];
      pk[3] = (__bf16)r3[j];
      *(v4bf*)&ldsT[(4 * cq + j) * 32 + 4 * p] = pk;
    }
    __syncthreads();
    // A: attn row m, two contiguous 8-bf16 runs (16B each)
    v8bf alo = *(const v8bf*)(arow + kc + 16 * g);
    v8bf ahi = *(const v8bf*)(arow + kc + 16 * g + 8);
    v16bf A;
#pragma unroll
    for (int i = 0; i < 8; ++i) { A[i] = alo[i]; A[8 + i] = ahi[i]; }
    // B: column (wave*16+m), rows 16g..16g+15 -> contiguous 32B in ldsT
    v16bf Bf = *(const v16bf*)&ldsT[(wave * 16 + m) * 32 + 16 * g];
    acc = __builtin_amdgcn_wmma_f32_16x16x32_bf16(false, A, false, Bf,
                                                  (short)0, acc, false, false);
  }
  // D: M = r + 8g (head), N = cbase + wave*16 + m (channel)
  const int col = cbase + wave * 16 + m;
#pragma unroll
  for (int r = 0; r < 8; ++r)
    atomicAdd(&t[((size_t)b * HEADS + (r + 8 * g)) * HIDDEN + col], acc[r]);
}

// ---------------- O1: x[b,hd] = t[b,h,:]·Wv[:,hd] + bv[hd] ------------------
__global__ void __launch_bounds__(256)
out_proj1(const float* __restrict__ t, const float* __restrict__ Wv,
          const float* __restrict__ bv, float* __restrict__ x) {
  const int gid = blockIdx.x * 256 + threadIdx.x;       // 16384
  const int b = gid >> 10, hd = gid & 1023, h = hd >> 6;
  const float* tp = t + ((size_t)(b * HEADS + h)) * HIDDEN;
  float acc = bv[hd];
  for (int c = 0; c < HIDDEN; ++c) acc += tp[c] * Wv[c * HIDDEN + hd];
  x[gid] = acc;
}

// ---------------- O2: out[b,j] = x[b,:]·Wo[:,j] + bo[j] ---------------------
__global__ void __launch_bounds__(256)
out_proj2(const float* __restrict__ x, const float* __restrict__ Wo,
          const float* __restrict__ bo, float* __restrict__ out) {
  const int gid = blockIdx.x * 256 + threadIdx.x;       // 16384
  const int b = gid >> 10, j = gid & 1023;
  const float* xp = x + b * HIDDEN;
  float acc = bo[j];
  for (int i = 0; i < HIDDEN; ++i) acc += xp[i] * Wo[i * HIDDEN + j];
  out[gid] = acc;
}

extern "C" void kernel_launch(void* const* d_in, const int* in_sizes, int n_in,
                              void* d_out, int out_size, void* d_ws, size_t ws_size,
                              hipStream_t stream) {
  const float* q    = (const float*)d_in[0];
  const float* kmat = (const float*)d_in[1];
  const float* vmat = (const float*)d_in[2];
  // d_in[3] = index_sample : provably dead for L_Q == 1, ignored.
  const float* Wq = (const float*)d_in[4];
  const float* bq = (const float*)d_in[5];
  const float* Wk = (const float*)d_in[6];
  const float* bk = (const float*)d_in[7];
  const float* Wv = (const float*)d_in[8];
  const float* bv = (const float*)d_in[9];
  const float* Wo = (const float*)d_in[10];
  const float* bo = (const float*)d_in[11];

  char* ws = (char*)d_ws;
  float*  qh    = (float*)(ws);                                   //  64 KB
  float*  sbias = (float*)(ws + (64 << 10));                      //   1 KB
  bf16_t* u     = (bf16_t*)(ws + (68 << 10));                     // 512 KB
  float*  s     = (float*)(ws + (640 << 10));                     //   4 MB
  bf16_t* attn  = (bf16_t*)(ws + (640 << 10) + (4 << 20));        //   2 MB
  float*  t     = (float*)(ws + (640 << 10) + (6 << 20));         //   1 MB
  float*  x     = (float*)(ws + (640 << 10) + (7 << 20));         //  64 KB

  prep_qh_zero_t<<<1024, 256, 0, stream>>>(q, Wq, bq, qh, t);
  prep_u        <<<1024, 256, 0, stream>>>(Wk, bk, qh, u, sbias);
  scores_kernel <<<dim3(16, 32), 256, 0, stream>>>(kmat, u, sbias, s);
  softmax_kernel<<<256, 256, 0, stream>>>(s, attn);
  av_kernel     <<<dim3(16, 8, 4), 256, 0, stream>>>(attn, vmat, t);
  out_proj1     <<<64, 256, 0, stream>>>(t, Wv, bv, x);
  out_proj2     <<<64, 256, 0, stream>>>(x, Wo, bo, (float*)d_out);
}
